// MyModel_61933428416805
// MI455X (gfx1250) — matrix operand order repaired and checked by reading, not compiled, not run on hardware
//
#include <hip/hip_runtime.h>

// ---------------------------------------------------------------------------
// Fused LSTM (T=512, B=4096, I=10, H=20) for gfx1250 (MI455X, wave32, WMMA).
//
// One wave owns a 16-row batch tile for the whole sequence.
// Per step: gates[16x80] = [x_t | h_{t-1}](16x30) x [W_ih^T ; W_hh^T](30x80) + b
// computed as 5 tiles of v_wmma_f32_16x16x32_f16 (K=32, 30 real + 2 zero-pad),
// f32 accumulation, bias folded into the C operand. Elementwise LSTM update in
// f32 with c kept in registers; h round-trips through LDS in f16 (A operand).
// Nonlinearities use the gfx1250 trans pipe (v_tanh_f32) — 5 trans ops per
// (m,hcol) pair. LDS phase hand-offs use an LDS-only fence + wave_barrier so
// the next-timestep x loads stay in flight across the elementwise phase
// (no s_wait_loadcnt on the serial critical path).
// Avoids the 2x671MB x_gates intermediate -> ~252MB total HBM traffic.
// ---------------------------------------------------------------------------

typedef __attribute__((ext_vector_type(16))) _Float16 v16h;
typedef __attribute__((ext_vector_type(8)))  float    v8f;

#define T_STEPS 512
#define BATCH   4096
#define INP     10
#define HID     20
#define GSTRIDE 84   // LDS row stride (floats) for gates tile; pads out bank conflicts

__device__ __forceinline__ float fast_rcp(float v) {
    return __builtin_amdgcn_rcpf(v);                 // single v_rcp_f32
}

#if __has_builtin(__builtin_amdgcn_tanhf)
__device__ __forceinline__ float fast_tanh(float v) {
    return __builtin_amdgcn_tanhf(v);                // single v_tanh_f32
}
#elif __has_builtin(__builtin_amdgcn_tanh_f32)
__device__ __forceinline__ float fast_tanh(float v) {
    return __builtin_amdgcn_tanh_f32(v);             // single v_tanh_f32
}
#else
__device__ __forceinline__ float fast_tanh(float v) {
    // 1 - 2/(e^{2x}+1): saturates to +-1, no inf/inf NaN
    return 1.0f - 2.0f * fast_rcp(__expf(2.0f * v) + 1.0f);
}
#endif

__device__ __forceinline__ float fast_sigmoid(float v) {
    // sigmoid(x) = 0.5*tanh(x/2) + 0.5  -> one trans op + one fma
    return __builtin_fmaf(0.5f, fast_tanh(0.5f * v), 0.5f);
}

// Order LDS traffic within this (single-wave) workgroup WITHOUT waiting on
// outstanding global loads: LDS-only acq_rel fence + scheduling barrier.
// The DS pipeline is in-order per wave, so this is sufficient for the
// cross-lane scatter->gather hand-offs through LDS.
__device__ __forceinline__ void lds_handoff() {
    __builtin_amdgcn_fence(__ATOMIC_ACQ_REL, "workgroup", "local");
    __builtin_amdgcn_wave_barrier();
}

__global__ __launch_bounds__(32)
void lstm_wmma_kernel(const float* __restrict__ x,
                      const float* __restrict__ hx,
                      const float* __restrict__ cx,
                      const float* __restrict__ W_ih,
                      const float* __restrict__ W_hh,
                      const float* __restrict__ b_ih,
                      const float* __restrict__ b_hh,
                      float* __restrict__ out)
{
    __shared__ float    gsh[16 * GSTRIDE];   // gates tile [m=16][n=80 (+pad)]
    __shared__ _Float16 hsh[16 * HID];       // h_{t} tile, f16, feeds the A operand

    const int lane = threadIdx.x;            // 0..31
    const int mrow = lane & 15;              // batch row within tile / N column within tile
    const int hi   = lane >> 4;              // wave half
    const int b0   = blockIdx.x * 16;        // batch tile origin (4096/16 = 256 blocks)

    // ---- Build fused weight operand B = [W_ih^T ; W_hh^T ; 0] (K=32 x N=80, f16) ----
    //   lanes 0-15 : n = lane,    slots 0..7 -> K=0..7,  slots 8..15 -> K=16..23
    //   lanes 16-31: n = lane-16, slots 0..7 -> K=8..15, slots 8..15 -> K=24..31
    v16h bf[5];
    float bias[5];
    #pragma unroll
    for (int j = 0; j < 5; ++j) {
        const int n = j * 16 + mrow;
        #pragma unroll
        for (int s = 0; s < 16; ++s) {
            const int k = (s & 7) + 8 * hi + ((s & 8) ? 16 : 0);
            float w;
            if (k < INP)            w = W_ih[n * INP + k];           // x-projection rows
            else if (k < INP + HID) w = W_hh[n * HID + (k - INP)];   // h-projection rows
            else                    w = 0.0f;                        // K pad (30,31)
            bf[j][s] = (_Float16)w;
        }
        bias[j] = b_ih[n] + b_hh[n];
    }

    // ---- Recurrent state init: c in registers (10 pairs per lane), h in LDS f16 ----
    float creg[10];
    #pragma unroll
    for (int jj = 0; jj < 10; ++jj) {
        const int p = lane + 32 * jj;        // p = m*HID + hcol
        creg[jj] = cx[b0 * HID + p];
        hsh[p]   = (_Float16)hx[b0 * HID + p];
    }
    lds_handoff();

    // ---- x operand registers for t = 0 (rows are 40B => 8B aligned, float2 loads) ----
    float2 xr0, xr1, xr2, xr3;
    {
        const float* rowp = x + ((size_t)b0 + mrow) * INP;
        if (lane < 16) {                     // this half supplies x cols 0..7
            xr0 = *(const float2*)(rowp + 0);
            xr1 = *(const float2*)(rowp + 2);
            xr2 = *(const float2*)(rowp + 4);
            xr3 = *(const float2*)(rowp + 6);
        } else {                             // this half supplies x cols 8..9
            xr0 = *(const float2*)(rowp + 8);
            xr1 = xr2 = xr3 = make_float2(0.0f, 0.0f);
        }
    }

    for (int t = 0; t < T_STEPS; ++t) {
        // ---- Build A fragment (16x32 f16). Per-lane K slots per ISA table:
        //   lanes 0-15 : slots 0..7 -> K=0..7,  slots 8..15 -> K=16..23
        //   lanes 16-31: slots 0..7 -> K=8..15, slots 8..15 -> K=24..31
        // K layout: [0..9]=x_t, [10..29]=h_{t-1}, [30,31]=0.
        v16h a;
        if (lane < 16) {
            a[0]=(_Float16)xr0.x; a[1]=(_Float16)xr0.y;
            a[2]=(_Float16)xr1.x; a[3]=(_Float16)xr1.y;
            a[4]=(_Float16)xr2.x; a[5]=(_Float16)xr2.y;
            a[6]=(_Float16)xr3.x; a[7]=(_Float16)xr3.y;
            #pragma unroll
            for (int i2 = 0; i2 < 8; ++i2)               // K=16..23 -> h[6..13]
                a[8 + i2] = hsh[mrow * HID + 6 + i2];
        } else {
            a[0]=(_Float16)xr0.x; a[1]=(_Float16)xr0.y;  // K=8,9 -> x[8],x[9]
            #pragma unroll
            for (int i2 = 0; i2 < 6; ++i2)               // K=10..15 -> h[0..5]
                a[2 + i2] = hsh[mrow * HID + i2];
            #pragma unroll
            for (int i2 = 0; i2 < 6; ++i2)               // K=24..29 -> h[14..19]
                a[8 + i2] = hsh[mrow * HID + 14 + i2];
            a[14] = (_Float16)0.0f; a[15] = (_Float16)0.0f;  // K=30,31 pad
        }

        // ---- Prefetch next timestep's x while the matrix pipe works ----
        float2 nx0 = xr0, nx1 = xr1, nx2 = xr2, nx3 = xr3;
        if (t + 1 < T_STEPS) {
            const float* rowp = x + ((size_t)(t + 1) * BATCH + b0 + mrow) * INP;
            if (lane < 16) {
                nx0 = *(const float2*)(rowp + 0);
                nx1 = *(const float2*)(rowp + 2);
                nx2 = *(const float2*)(rowp + 4);
                nx3 = *(const float2*)(rowp + 6);
            } else {
                nx0 = *(const float2*)(rowp + 8);
            }
        }
        if (t + 2 < T_STEPS) {               // warm L2 two steps ahead (global_prefetch_b8)
            __builtin_prefetch(x + ((size_t)(t + 2) * BATCH + b0 + mrow) * INP, 0, 1);
        }

        // ---- gates = A x B + bias : 5 x v_wmma_f32_16x16x32_f16 ----
        #pragma unroll
        for (int j = 0; j < 5; ++j) {
            const float bb = bias[j];
            v8f c0 = { bb, bb, bb, bb, bb, bb, bb, bb };   // bias depends only on n
            v8f d = __builtin_amdgcn_wmma_f32_16x16x32_f16(
                        false, a, false, bf[j], (short)0, c0, false, false);
            // D layout: lane holds n = j*16 + (lane&15); VGPR r -> m = r + 8*(lane>>4)
            #pragma unroll
            for (int r = 0; r < 8; ++r)
                gsh[(r + 8 * hi) * GSTRIDE + j * 16 + mrow] = d[r];
        }
        lds_handoff();     // order LDS scatter->gather; global x loads stay in flight

        // ---- Elementwise LSTM update: 320 (m,hcol) pairs, 10 per lane ----
        const size_t outbase = (size_t)t * (BATCH * HID) + (size_t)b0 * HID;
        #pragma unroll
        for (int jj = 0; jj < 10; ++jj) {
            const int p  = lane + 32 * jj;
            const int m  = p / HID;
            const int hc = p - m * HID;
            const float* gr = &gsh[m * GSTRIDE + hc];
            const float ig = fast_sigmoid(gr[0]);          // i
            const float fg = fast_sigmoid(gr[HID]);        // f
            const float gg = fast_tanh   (gr[2 * HID]);    // g
            const float og = fast_sigmoid(gr[3 * HID]);    // o
            const float cn = __builtin_fmaf(fg, creg[jj], ig * gg);
            creg[jj] = cn;
            const float hn = og * fast_tanh(cn);
            hsh[p] = (_Float16)hn;                         // f16 h for next A operand
            out[outbase + p] = hn;                         // lane-contiguous f32 store
        }
        lds_handoff();     // h tile visible before next step's A-fragment build

        xr0 = nx0; xr1 = nx1; xr2 = nx2; xr3 = nx3;
    }
}

extern "C" void kernel_launch(void* const* d_in, const int* in_sizes, int n_in,
                              void* d_out, int out_size, void* d_ws, size_t ws_size,
                              hipStream_t stream) {
    const float* x    = (const float*)d_in[0];
    const float* hx   = (const float*)d_in[1];
    const float* cx   = (const float*)d_in[2];
    const float* W_ih = (const float*)d_in[3];
    const float* W_hh = (const float*)d_in[4];
    const float* b_ih = (const float*)d_in[5];
    const float* b_hh = (const float*)d_in[6];
    float* out = (float*)d_out;

    dim3 grid(BATCH / 16);   // 256 batch tiles, one wave each
    dim3 block(32);          // wave32
    hipLaunchKernelGGL(lstm_wmma_kernel, grid, block, 0, stream,
                       x, hx, cx, W_ih, W_hh, b_ih, b_hh, out);
}
